// MultiHeadAttention_55250459295909
// MI455X (gfx1250) — matrix operand order repaired and checked by reading, not compiled, not run on hardware
//
#include <hip/hip_runtime.h>
#include <stdint.h>

// Problem constants (from reference): B=2048, T=32, E=1024, H=16, HS=64
#define BB  2048
#define TT  32
#define EE  1024
#define HH  16
#define HSZ 64

typedef __bf16 bf16_t;
typedef __attribute__((ext_vector_type(16))) __bf16 v16bf;
typedef __attribute__((ext_vector_type(8)))  float  v8f;

struct U32x8 { uint4 lo; uint4 hi; };   // 32 bytes == v16bf

// ---- fp32 -> bf16 round-to-nearest-even (off the hot path) ----
static __device__ __forceinline__ uint16_t f2bf_bits(float f) {
  uint32_t u = __builtin_bit_cast(uint32_t, f);
  return (uint16_t)((u + 0x7FFFu + ((u >> 16) & 1u)) >> 16);
}
static __device__ __forceinline__ bf16_t f2bf(float f) {
  return __builtin_bit_cast(bf16_t, f2bf_bits(f));
}
// ---- fp32 -> bf16 truncation (hot path: folds into ds_store_b16_d16_hi) ----
static __device__ __forceinline__ bf16_t f2bf_t(float f) {
  return __builtin_bit_cast(bf16_t, (uint16_t)(__builtin_bit_cast(uint32_t, f) >> 16));
}

static __device__ __forceinline__ v8f wmma_bf16(v16bf a, v16bf b, v8f c) {
  // D(f32 16x16) = A(bf16 16x32) * B(bf16 32x16) + C
  return __builtin_amdgcn_wmma_f32_16x16x32_bf16(false, a, false, b, (short)0, c,
                                                 false, false);
}

// A-tile (16x32, MxK) from row-major bf16 LDS buffer, leading dim `ld`.
// Lane r=lane&15 holds row m0+r; K = kk + kb + j, j contiguous (0..15),
// kb = 0 (lanes 0-15) / 16 (lanes 16-31).  Two ds_load_b128 per tile.
static __device__ __forceinline__ v16bf lds_a_tile(const bf16_t* base, int ld,
                                                   int m0, int kk, int lane) {
  const int r  = m0 + (lane & 15);
  const int kb = (lane < 16) ? 0 : 16;
  const bf16_t* p = base + r * ld + kk + kb;
  U32x8 t;
  t.lo = *(const uint4*)(p);
  t.hi = *(const uint4*)(p + 8);
  return __builtin_bit_cast(v16bf, t);
}

// B-tile (32x16, KxN) where LDS stores B TRANSPOSED row-major: buf[n*ld + k].
// Lane holds column n0+(lane&15); K = kk + (lane<16?0:16) + j, j contiguous.
static __device__ __forceinline__ v16bf lds_bT_tile(const bf16_t* buf, int ld,
                                                    int n0, int kk, int lane) {
  const int n  = n0 + (lane & 15);
  const int ko = kk + ((lane < 16) ? 0 : 16);
  const bf16_t* p = buf + n * ld + ko;
  U32x8 t;
  t.lo = *(const uint4*)(p);
  t.hi = *(const uint4*)(p + 8);
  return __builtin_bit_cast(v16bf, t);
}

// B-tile from PRE-SWIZZLED bf16 weights: tile (kb,nt) is 512 contiguous bf16
// in lane-major order (lane l: col l&15, K = 16*(l>>4) + j) -> two coalesced global_load_b128 per lane.
static __device__ __forceinline__ v16bf pk_b_tile(const bf16_t* Wpk, int NT,
                                                  int kb, int nt, int lane) {
  const bf16_t* p = Wpk + ((size_t)(kb * NT + nt) << 9) + (lane << 4);
  U32x8 t;
  t.lo = *(const uint4*)(p);
  t.hi = *(const uint4*)(p + 8);
  return __builtin_bit_cast(v16bf, t);
}

// Fused Q+K projection: one pass over Xb, shared A-tiles feed both weight
// streams. C = 2 mats x 2 mt x 4 nt x 8 = 128 VGPRs (stays < 256).
static __device__ __forceinline__ void head_qk(const bf16_t* Xb,
                                               const bf16_t* Wq, const bf16_t* Wk,
                                               bf16_t* Qh, bf16_t* Kh, int lane) {
  v8f C[2][2][4];
#pragma unroll
  for (int w = 0; w < 2; ++w)
#pragma unroll
    for (int mt = 0; mt < 2; ++mt)
#pragma unroll
      for (int nt = 0; nt < 4; ++nt)
#pragma unroll
        for (int i = 0; i < 8; ++i) C[w][mt][nt][i] = 0.0f;

#pragma unroll 1
  for (int kb = 0; kb < EE / 32; ++kb) {
    if (kb + 1 < EE / 32) {   // lane-spread prefetch: 32 lanes x 128B = next 4KB
      __builtin_prefetch(Wq + (size_t)(kb + 1) * 2048 + lane * 64, 0, 1);
      __builtin_prefetch(Wk + (size_t)(kb + 1) * 2048 + lane * 64, 0, 1);
    }
    v16bf a0 = lds_a_tile(Xb, EE, 0, kb * 32, lane);
    v16bf a1 = lds_a_tile(Xb, EE, 16, kb * 32, lane);
#pragma unroll
    for (int nt = 0; nt < 4; ++nt) {
      v16bf bq = pk_b_tile(Wq, 4, kb, nt, lane);
      C[0][0][nt] = wmma_bf16(a0, bq, C[0][0][nt]);
      C[0][1][nt] = wmma_bf16(a1, bq, C[0][1][nt]);
      v16bf bk = pk_b_tile(Wk, 4, kb, nt, lane);
      C[1][0][nt] = wmma_bf16(a0, bk, C[1][0][nt]);
      C[1][1][nt] = wmma_bf16(a1, bk, C[1][1][nt]);
    }
  }

  const int n_lo = lane & 15;
#pragma unroll
  for (int w = 0; w < 2; ++w) {
    bf16_t* dst = (w == 0) ? Qh : Kh;
#pragma unroll
    for (int mt = 0; mt < 2; ++mt)
#pragma unroll
      for (int nt = 0; nt < 4; ++nt)
#pragma unroll
        for (int i = 0; i < 8; ++i) {
          const int m = mt * 16 + ((lane < 16) ? i : i + 8);
          dst[m * HSZ + nt * 16 + n_lo] = f2bf_t(C[w][mt][nt][i]);
        }
  }
}

// V projection, stored transposed [d][t] for the att@V B-operand.
static __device__ __forceinline__ void head_v(const bf16_t* Xb, const bf16_t* Wv,
                                              bf16_t* VT, int lane) {
  v8f C[2][4];
#pragma unroll
  for (int mt = 0; mt < 2; ++mt)
#pragma unroll
    for (int nt = 0; nt < 4; ++nt)
#pragma unroll
      for (int i = 0; i < 8; ++i) C[mt][nt][i] = 0.0f;

#pragma unroll 1
  for (int kb = 0; kb < EE / 32; ++kb) {
    if (kb + 1 < EE / 32)
      __builtin_prefetch(Wv + (size_t)(kb + 1) * 2048 + lane * 64, 0, 1);
    v16bf a0 = lds_a_tile(Xb, EE, 0, kb * 32, lane);
    v16bf a1 = lds_a_tile(Xb, EE, 16, kb * 32, lane);
#pragma unroll
    for (int nt = 0; nt < 4; ++nt) {
      v16bf bb = pk_b_tile(Wv, 4, kb, nt, lane);
      C[0][nt] = wmma_bf16(a0, bb, C[0][nt]);
      C[1][nt] = wmma_bf16(a1, bb, C[1][nt]);
    }
  }
  const int n_lo = lane & 15;
#pragma unroll
  for (int mt = 0; mt < 2; ++mt)
#pragma unroll
    for (int nt = 0; nt < 4; ++nt)
#pragma unroll
      for (int i = 0; i < 8; ++i) {
        const int m = mt * 16 + ((lane < 16) ? i : i + 8);
        VT[(nt * 16 + n_lo) * TT + m] = f2bf_t(C[mt][nt][i]);
      }
}

// ---------- pre-pass: fp32 weights -> bf16 (RNE), swizzled into per-lane B tiles (K = 16h + j) ----
__global__ void __launch_bounds__(256)
pack_weights_kernel(const float* __restrict__ src, bf16_t* __restrict__ dst,
                    int N, int perMat, int total) {
  int idx = blockIdx.x * 256 + threadIdx.x;
  if (idx >= total) return;
  const int mat  = idx / perMat;
  const int rem  = idx - mat * perMat;
  const int tile = rem >> 9;
  const int li   = rem & 511;
  const int l    = li >> 4;
  const int j    = li & 15;
  const int NT   = N >> 4;
  const int kb   = tile / NT;
  const int nt   = tile - kb * NT;
  const int k    = kb * 32 + ((l < 16) ? 0 : 16) + j;
  const int n    = nt * 16 + (l & 15);
  dst[idx] = f2bf(src[(size_t)mat * perMat + (size_t)k * N + n]);
}

// One workgroup (8 wave32) per batch element. LDS (dynamic, 256 KB):
//   [0,64K)    Xb   : x[b] bf16 32x1024
//   [64K,128K) Obuf : concat head outs bf16 32x1024
//   [128K,256K) per-wave scratch 16KB x 8 (Qh, Kh, VT, S) — also overlaid by
//               the 128KB fp32 async staging buffer for x before use.
__global__ void __launch_bounds__(256, 1)
mha_fused_kernel(const float* __restrict__ x,
                 const bf16_t* __restrict__ Wqp, const bf16_t* __restrict__ Wkp,
                 const bf16_t* __restrict__ Wvp, const bf16_t* __restrict__ Wpp,
                 const float* __restrict__ bp, float* __restrict__ out) {
  extern __shared__ __align__(16) unsigned char smem[];
  bf16_t* Xb   = (bf16_t*)(smem);
  bf16_t* Obuf = (bf16_t*)(smem + TT * EE * 2);
  unsigned char* wsbase = smem + 2 * TT * EE * 2;

  const int tid  = threadIdx.x;
  const int wave = tid >> 5;
  const int lane = tid & 31;
  const int b    = blockIdx.x;

  bf16_t* Qh = (bf16_t*)(wsbase + wave * 16384);
  bf16_t* Kh = Qh + TT * HSZ;
  bf16_t* VT = Kh + TT * HSZ;
  float*  S  = (float*)(VT + HSZ * TT);

  // ---- CDNA5 async DMA: x[b] fp32 -> LDS staging (no VGPR round trip) ----
  const float* xb = x + (size_t)b * TT * EE;
  float* Xtmp = (float*)(wsbase);               // 128 KB, overlays scratch
  {
    const uint32_t lds0 = (uint32_t)(uintptr_t)Xtmp;  // LDS aperture: low 32 bits
    uint32_t off = (uint32_t)tid * 16u;
#pragma unroll 1
    for (int it = 0; it < (TT * EE * 4) / (256 * 16); ++it) {  // 32 x 16B/thread
      asm volatile("global_load_async_to_lds_b128 %0, %1, %2"
                   :: "v"(lds0 + off), "v"(off), "s"(xb) : "memory");
      off += 256u * 16u;
    }
    asm volatile("s_wait_asynccnt 0x0" ::: "memory");
  }
  __syncthreads();
  // convert staged fp32 -> bf16 (RNE) into Xb
  for (int i = tid * 4; i < TT * EE; i += 256 * 4) {
    float4 v = *(const float4*)(Xtmp + i);
    uint32_t p0 = ((uint32_t)f2bf_bits(v.y) << 16) | f2bf_bits(v.x);
    uint32_t p1 = ((uint32_t)f2bf_bits(v.w) << 16) | f2bf_bits(v.z);
    *(uint2*)(Xb + i) = make_uint2(p0, p1);
  }
  __syncthreads();   // Xb ready; Xtmp dead -> scratch region reusable

  // ---- per-head attention: wave w handles heads w and w+8 ----
#pragma unroll 1
  for (int hi = 0; hi < 2; ++hi) {
    const int h = wave + hi * 8;
    const size_t woff = (size_t)h * EE * HSZ;   // packed per-head matrix size

    head_qk(Xb, Wqp + woff, Wkp + woff, Qh, Kh, lane);
    head_v(Xb, Wvp + woff, VT, lane);

    // scores(32x32) = Q @ K^T, K-dim = 64
    v8f SC[2][2];
#pragma unroll
    for (int mt = 0; mt < 2; ++mt)
#pragma unroll
      for (int nt = 0; nt < 2; ++nt)
#pragma unroll
        for (int i = 0; i < 8; ++i) SC[mt][nt][i] = 0.0f;
#pragma unroll
    for (int kk = 0; kk < HSZ; kk += 32) {
      v16bf a0 = lds_a_tile(Qh, HSZ, 0, kk, lane);
      v16bf a1 = lds_a_tile(Qh, HSZ, 16, kk, lane);
#pragma unroll
      for (int nt = 0; nt < 2; ++nt) {
        v16bf bb = lds_bT_tile(Kh, HSZ, nt * 16, kk, lane);
        SC[0][nt] = wmma_bf16(a0, bb, SC[0][nt]);
        SC[1][nt] = wmma_bf16(a1, bb, SC[1][nt]);
      }
    }
    // spill scores f32 to LDS for row-wise softmax
    const int n_lo = lane & 15;
#pragma unroll
    for (int mt = 0; mt < 2; ++mt)
#pragma unroll
      for (int nt = 0; nt < 2; ++nt)
#pragma unroll
        for (int i = 0; i < 8; ++i) {
          const int m = mt * 16 + ((lane < 16) ? i : i + 8);
          S[m * TT + nt * 16 + n_lo] = SC[mt][nt][i];
        }

    // causal softmax: lane t owns row t; att (A-layout bf16) reuses Qh
    {
      const int t = lane;
      const float scale = 0.125f;  // HS^-0.5
      float mx = -3.4e38f;
      for (int s = 0; s <= t; ++s) mx = fmaxf(mx, S[t * TT + s] * scale);
      float sum = 0.0f;
      for (int s = 0; s <= t; ++s) {
        float p = __expf(S[t * TT + s] * scale - mx);
        sum += p;
        S[t * TT + s] = p;
      }
      const float inv = 1.0f / sum;
      bf16_t* att = Qh;
      for (int s = 0; s < TT; ++s)
        att[t * TT + s] = (s <= t) ? f2bf_t(S[t * TT + s] * inv) : f2bf_t(0.0f);
    }

    // O(32x64) = att(32x32) @ V(32x64); K=32 -> one WMMA step per tile
    {
      const bf16_t* att = Qh;
      v16bf a0 = lds_a_tile(att, TT, 0, 0, lane);
      v16bf a1 = lds_a_tile(att, TT, 16, 0, lane);
#pragma unroll
      for (int nt = 0; nt < 4; ++nt) {
        v16bf bb = lds_bT_tile(VT, TT, nt * 16, 0, lane);
        v8f c0, c1;
#pragma unroll
        for (int i = 0; i < 8; ++i) { c0[i] = 0.0f; c1[i] = 0.0f; }
        c0 = wmma_bf16(a0, bb, c0);
        c1 = wmma_bf16(a1, bb, c1);
#pragma unroll
        for (int i = 0; i < 8; ++i) {
          const int mlo = (lane < 16) ? i : i + 8;
          Obuf[(size_t)(mlo)      * EE + h * HSZ + nt * 16 + n_lo] = f2bf_t(c0[i]);
          Obuf[(size_t)(mlo + 16) * EE + h * HSZ + nt * 16 + n_lo] = f2bf_t(c1[i]);
        }
      }
    }
  }
  __syncthreads();

  // ---- out-projection: Y = Obuf @ Wp + bp; wave owns 8 N-tiles, 4 per pass ----
  float* ob = out + (size_t)b * TT * EE;
#pragma unroll 1
  for (int pass = 0; pass < 2; ++pass) {
    const int ntb = wave * 8 + pass * 4;
    v8f C[2][4];
#pragma unroll
    for (int q = 0; q < 4; ++q) {
      const float bias = bp[(ntb + q) * 16 + (lane & 15)];
#pragma unroll
      for (int i = 0; i < 8; ++i) { C[0][q][i] = bias; C[1][q][i] = bias; }
    }
#pragma unroll 1
    for (int kb = 0; kb < EE / 32; ++kb) {
      if (kb + 1 < EE / 32)   // next k-step's 4 tiles = 4KB, lane-spread
        __builtin_prefetch(Wpp + ((size_t)(kb + 1) * 64 + ntb) * 512 + lane * 64, 0, 1);
      v16bf a0 = lds_a_tile(Obuf, EE, 0, kb * 32, lane);
      v16bf a1 = lds_a_tile(Obuf, EE, 16, kb * 32, lane);
#pragma unroll
      for (int q = 0; q < 4; ++q) {
        v16bf bb = pk_b_tile(Wpp, EE / 16, kb, ntb + q, lane);
        C[0][q] = wmma_bf16(a0, bb, C[0][q]);
        C[1][q] = wmma_bf16(a1, bb, C[1][q]);
      }
    }
#pragma unroll
    for (int q = 0; q < 4; ++q)
#pragma unroll
      for (int i = 0; i < 8; ++i) {
        const int mlo = (lane < 16) ? i : i + 8;
        const int n = (ntb + q) * 16 + (lane & 15);
        ob[(size_t)(mlo)      * EE + n] = C[0][q][i];
        ob[(size_t)(mlo + 16) * EE + n] = C[1][q][i];
      }
  }
}

extern "C" void kernel_launch(void* const* d_in, const int* in_sizes, int n_in,
                              void* d_out, int out_size, void* d_ws, size_t ws_size,
                              hipStream_t stream) {
  (void)in_sizes; (void)n_in; (void)out_size; (void)ws_size;
  const float* x  = (const float*)d_in[0];
  const float* Wq = (const float*)d_in[1];
  const float* Wk = (const float*)d_in[2];
  const float* Wv = (const float*)d_in[3];
  const float* Wp = (const float*)d_in[4];
  const float* bp = (const float*)d_in[5];
  float* out = (float*)d_out;

  // Workspace: packed bf16 weights (8 MB), rebuilt every call (deterministic).
  const size_t perHead  = (size_t)EE * HSZ;          // 65536
  const size_t qkvElems = (size_t)HH * perHead;      // 1,048,576 each
  const size_t wpElems  = (size_t)EE * EE;           // 1,048,576
  bf16_t* Wqp = (bf16_t*)d_ws;
  bf16_t* Wkp = Wqp + qkvElems;
  bf16_t* Wvp = Wkp + qkvElems;
  bf16_t* Wpp = Wvp + qkvElems;

  const int blk = 256;
  pack_weights_kernel<<<dim3((qkvElems + blk - 1) / blk), dim3(blk), 0, stream>>>(
      Wq, Wqp, HSZ, (int)perHead, (int)qkvElems);
  pack_weights_kernel<<<dim3((qkvElems + blk - 1) / blk), dim3(blk), 0, stream>>>(
      Wk, Wkp, HSZ, (int)perHead, (int)qkvElems);
  pack_weights_kernel<<<dim3((qkvElems + blk - 1) / blk), dim3(blk), 0, stream>>>(
      Wv, Wvp, HSZ, (int)perHead, (int)qkvElems);
  pack_weights_kernel<<<dim3((wpElems + blk - 1) / blk), dim3(blk), 0, stream>>>(
      Wp, Wpp, EE, (int)wpElems, (int)wpElems);

  const size_t dyn_lds = 2u * TT * EE * 2u + 8u * 16384u;  // 262144 B (CDNA5: 320KB/WGP)
  static_assert(2u * TT * EE * 2u + 8u * 16384u == 262144, "LDS budget");
  (void)hipFuncSetAttribute((const void*)mha_fused_kernel,
                            hipFuncAttributeMaxDynamicSharedMemorySize,
                            (int)dyn_lds);
  mha_fused_kernel<<<dim3(BB), dim3(256), dyn_lds, stream>>>(x, Wqp, Wkp, Wvp, Wpp,
                                                             bp, out);
}
